// GNN_11965778887059
// MI455X (gfx1250) — matrix operand order repaired and checked by reading, not compiled, not run on hardware
//
#include <hip/hip_runtime.h>

// ---------------------------------------------------------------------------
// GCNConv on a fully connected graph (N=4096 nodes, F=2048 feats), bf16 WMMA.
//   d[j]   = rsqrt(sum_i ew[i*N+j])          (0 if deg==0)
//   h      = x @ W                           (bf16 WMMA, f32 acc, stored bf16)
//   out    = d[j] * sum_i (ew[i*N+j]*d[i]) * h[i,f] + b[f]
// Workspace layout: [deg f32 16KB][dvec f32 16KB][h bf16 16MB]
// Pipelined: global loads for slab k+1 issue before WMMAs of slab k;
// double-buffered LDS -> one barrier per K iteration. Transposed staging
// packs 4 K-consecutive bf16 per ds_store_b64 (no scalar b16 stores).
// ---------------------------------------------------------------------------

typedef __attribute__((ext_vector_type(16))) __bf16          v16bf;
typedef __attribute__((ext_vector_type(8)))  float           v8f;
typedef __attribute__((ext_vector_type(8)))  unsigned short  us8;
typedef __attribute__((ext_vector_type(4)))  unsigned short  us4;

union FragBF {
  us8   h[2];
  v16bf v;
};

__device__ __forceinline__ unsigned short f2bf(float f) {
  __bf16 b = (__bf16)f;                 // hardware v_cvt_pk_bf16_f32 path
  return __builtin_bit_cast(unsigned short, b);
}

// ---------------- degree / normalization ----------------------------------

__global__ void GNN_deg_init(float* __restrict__ deg, int n) {
  int j = blockIdx.x * blockDim.x + threadIdx.x;
  if (j < n) deg[j] = 0.0f;
}

__global__ void GNN_deg_partial(const float* __restrict__ ew,
                                float* __restrict__ deg, int n) {
  int j  = blockIdx.x * 256 + threadIdx.x;
  int i0 = blockIdx.y * 128;
  float s = 0.0f;
#pragma unroll 4
  for (int i = 0; i < 128; ++i)
    s += ew[(size_t)(i0 + i) * n + j];   // coalesced across j
  atomicAdd(&deg[j], s);
}

__global__ void GNN_deg_finalize(const float* __restrict__ deg,
                                 float* __restrict__ dvec, int n) {
  int j = blockIdx.x * blockDim.x + threadIdx.x;
  if (j < n) {
    float s = deg[j];
    dvec[j] = (s > 0.0f) ? rsqrtf(s) : 0.0f;
  }
}

// ---------------- per-wave 64x32 WMMA over one K=32 LDS slab ---------------
// lds_a: [128][32] bf16, row-major (M x K)
// lds_b: [128][32] bf16, "N-major" (N x K)  -> B-frag K contiguous per lane
__device__ __forceinline__ void wave_mma32(const unsigned short* lds_a,
                                           const unsigned short* lds_b,
                                           int wave_m, int wave_n, int lane,
                                           v8f acc[4][2]) {
  const int l16 = lane >> 4;
  const int lm  = lane & 15;
  FragBF a[4], b[2];
#pragma unroll
  for (int mt = 0; mt < 4; ++mt) {
    const int m = wave_m * 64 + mt * 16 + lm;
    // ISA 16-bit A 16x32 layout: lane half l16 holds K = l16*8..+7 and 16+l16*8..+7
    a[mt].h[0] = *(const us8*)&lds_a[m * 32 + l16 * 8];
    a[mt].h[1] = *(const us8*)&lds_a[m * 32 + 16 + l16 * 8];
  }
#pragma unroll
  for (int nt = 0; nt < 2; ++nt) {
    const int nn = wave_n * 32 + nt * 16 + lm;
    // ISA 16-bit B 32x16 layout: lane half l16 holds K = l16*8..+7 and 16+l16*8..+7 for col lm
    b[nt].h[0] = *(const us8*)&lds_b[nn * 32 + l16 * 8];
    b[nt].h[1] = *(const us8*)&lds_b[nn * 32 + 16 + l16 * 8];
  }
#pragma unroll
  for (int mt = 0; mt < 4; ++mt)
#pragma unroll
    for (int nt = 0; nt < 2; ++nt)
      acc[mt][nt] = __builtin_amdgcn_wmma_f32_16x16x32_bf16(
          false, a[mt].v, false, b[nt].v, (short)0, acc[mt][nt], false, false);
}

// ---------------- GEMM1: h = x @ W  (f32 in, bf16 out) ---------------------

__global__ __launch_bounds__(256)
void GNN_gemm_xw(const float* __restrict__ X, const float* __restrict__ W,
                 unsigned short* __restrict__ Hbf, int M, int N, int K) {
  __shared__ unsigned short lds_a[2][128 * 32];
  __shared__ unsigned short lds_b[2][128 * 32];
  const int tid = threadIdx.x;
  const int wid = tid >> 5, lane = tid & 31;
  const int wave_m = wid >> 2, wave_n = wid & 3;
  const int m0 = blockIdx.y * 128, n0 = blockIdx.x * 128;
  const int r  = tid >> 3, c = (tid & 7) * 4;    // A staging: row r(+p*32), cols c..c+3
  const int kb = (tid >> 5) * 4;                 // B staging: K rows kb..kb+3
  const int f4 = (tid & 31) * 4;                 //            cols f4..f4+3

  v8f acc[4][2];
#pragma unroll
  for (int mt = 0; mt < 4; ++mt)
#pragma unroll
    for (int nt = 0; nt < 2; ++nt)
#pragma unroll
      for (int e = 0; e < 8; ++e) acc[mt][nt][e] = 0.0f;

  float4 ra[4];
  float  rb[4][4];                                // [k in kb..kb+3][f in f4..f4+3]
  auto loadT = [&](int k0) {
#pragma unroll
    for (int p = 0; p < 4; ++p)
      ra[p] = *(const float4*)&X[(size_t)(m0 + r + p * 32) * K + k0 + c];
#pragma unroll
    for (int p = 0; p < 4; ++p) {
      const float4 t = *(const float4*)&W[(size_t)(k0 + kb + p) * N + n0 + f4];
      rb[p][0] = t.x; rb[p][1] = t.y; rb[p][2] = t.z; rb[p][3] = t.w;
    }
  };
  auto storeT = [&](int buf) {
#pragma unroll
    for (int p = 0; p < 4; ++p) {      // A: contiguous along K -> merged stores
      const int row = r + p * 32;
      us4 v = {f2bf(ra[p].x), f2bf(ra[p].y), f2bf(ra[p].z), f2bf(ra[p].w)};
      *(us4*)&lds_a[buf][row * 32 + c] = v;
    }
#pragma unroll
    for (int q = 0; q < 4; ++q) {      // B transpose: pack 4 K-consecutive bf16
      us4 v = {f2bf(rb[0][q]), f2bf(rb[1][q]), f2bf(rb[2][q]), f2bf(rb[3][q])};
      *(us4*)&lds_b[buf][(f4 + q) * 32 + kb] = v;
    }
  };

  loadT(0);
  storeT(0);
  __syncthreads();
  for (int k0 = 0; k0 < K; k0 += 32) {
    const int  cur  = (k0 >> 5) & 1;
    const bool more = (k0 + 32) < K;
    if (more) loadT(k0 + 32);                         // loads in flight...
    if (k0 + 64 < K) {                                // ...and L2 warm-up
      __builtin_prefetch(&X[(size_t)(m0 + r) * K + k0 + 64 + c], 0, 0);
      __builtin_prefetch(&W[(size_t)(k0 + 64 + kb) * N + n0 + f4], 0, 0);
    }
    wave_mma32(lds_a[cur], lds_b[cur], wave_m, wave_n, lane, acc);
    if (more) storeT(cur ^ 1);                        // waits loads here
    __syncthreads();
  }

  // epilogue: f32 acc -> bf16 h
  const int l16 = lane >> 4, lm = lane & 15;
#pragma unroll
  for (int mt = 0; mt < 4; ++mt)
#pragma unroll
    for (int nt = 0; nt < 2; ++nt)
#pragma unroll
      for (int rr = 0; rr < 8; ++rr) {
        const int m  = m0 + wave_m * 64 + mt * 16 + (l16 ? rr + 8 : rr);
        const int ff = n0 + wave_n * 32 + nt * 16 + lm;
        Hbf[(size_t)m * N + ff] = f2bf(acc[mt][nt][rr]);
      }
}

// ---------------- GEMM2: out = diag(d) * A^T * diag(d) * h + b -------------
// A-operand tile AnT[j,i] = ew[i*N+j]*d[i]; transpose folded into LDS staging.

__global__ __launch_bounds__(256)
void GNN_gemm_agg(const float* __restrict__ ew,
                  const unsigned short* __restrict__ Hbf,
                  const float* __restrict__ dvec,
                  const float* __restrict__ bias,
                  float* __restrict__ out, int Nn, int F) {
  __shared__ unsigned short lds_a[2][128 * 32];
  __shared__ unsigned short lds_b[2][128 * 32];
  const int tid = threadIdx.x;
  const int wid = tid >> 5, lane = tid & 31;
  const int wave_m = wid >> 2, wave_n = wid & 3;
  const int j0 = blockIdx.y * 128, f0 = blockIdx.x * 128;
  const int kb = (tid >> 5) * 4;                 // K rows kb..kb+3 (i index)
  const int c4 = (tid & 31) * 4;                 // col group (j for A, f for B)

  v8f acc[4][2];
#pragma unroll
  for (int mt = 0; mt < 4; ++mt)
#pragma unroll
    for (int nt = 0; nt < 2; ++nt)
#pragma unroll
      for (int e = 0; e < 8; ++e) acc[mt][nt][e] = 0.0f;

  float          ra[4][4];   // ew micro-tile, d[i]-scaled at pack time
  float          dk4[4];
  unsigned short rb[4][4];   // h bf16 micro-tile
  auto loadT = [&](int i0) {
#pragma unroll
    for (int p = 0; p < 4; ++p) {
      dk4[p] = dvec[i0 + kb + p];
      const float4 t = *(const float4*)&ew[(size_t)(i0 + kb + p) * Nn + j0 + c4];
      ra[p][0] = t.x; ra[p][1] = t.y; ra[p][2] = t.z; ra[p][3] = t.w;
    }
#pragma unroll
    for (int p = 0; p < 4; ++p) {
      const us4 t = *(const us4*)&Hbf[(size_t)(i0 + kb + p) * F + f0 + c4];
      rb[p][0] = t.x; rb[p][1] = t.y; rb[p][2] = t.z; rb[p][3] = t.w;
    }
  };
  auto storeT = [&](int buf) {
#pragma unroll
    for (int q = 0; q < 4; ++q) {   // A^T: pack 4 K-consecutive scaled bf16
      us4 v = {f2bf(ra[0][q] * dk4[0]), f2bf(ra[1][q] * dk4[1]),
               f2bf(ra[2][q] * dk4[2]), f2bf(ra[3][q] * dk4[3])};
      *(us4*)&lds_a[buf][(c4 + q) * 32 + kb] = v;
    }
#pragma unroll
    for (int q = 0; q < 4; ++q) {   // B: pure ushort transpose shuffle
      us4 v = {rb[0][q], rb[1][q], rb[2][q], rb[3][q]};
      *(us4*)&lds_b[buf][(c4 + q) * 32 + kb] = v;
    }
  };

  loadT(0);
  storeT(0);
  __syncthreads();
  for (int i0 = 0; i0 < Nn; i0 += 32) {
    const int  cur  = (i0 >> 5) & 1;
    const bool more = (i0 + 32) < Nn;
    if (more) loadT(i0 + 32);
    if (i0 + 64 < Nn) {
      __builtin_prefetch(&ew[(size_t)(i0 + 64 + kb) * Nn + j0 + c4], 0, 0);
      __builtin_prefetch(&Hbf[(size_t)(i0 + 64 + kb) * F + f0 + c4], 0, 0);
    }
    wave_mma32(lds_a[cur], lds_b[cur], wave_m, wave_n, lane, acc);
    if (more) storeT(cur ^ 1);
    __syncthreads();
  }

  // epilogue: out[j,f] = d[j]*acc + b[f]
  const int l16 = lane >> 4, lm = lane & 15;
#pragma unroll
  for (int mt = 0; mt < 4; ++mt)
#pragma unroll
    for (int nt = 0; nt < 2; ++nt) {
      const int ff = f0 + wave_n * 32 + nt * 16 + lm;
      const float bf = bias[ff];
#pragma unroll
      for (int rr = 0; rr < 8; ++rr) {
        const int j = j0 + wave_m * 64 + mt * 16 + (l16 ? rr + 8 : rr);
        out[(size_t)j * F + ff] = dvec[j] * acc[mt][nt][rr] + bf;
      }
    }
}

// ---------------------------------------------------------------------------

extern "C" void kernel_launch(void* const* d_in, const int* in_sizes, int n_in,
                              void* d_out, int out_size, void* d_ws, size_t ws_size,
                              hipStream_t stream) {
  const float* x  = (const float*)d_in[0];
  // d_in[1] = edge_index: fully-connected meshgrid, e = i*n + j -> not needed
  const float* ew = (const float*)d_in[2];
  const float* W  = (const float*)d_in[3];
  const float* b  = (const float*)d_in[4];
  float* out = (float*)d_out;

  const int n = 4096;   // N_CHANS
  const int F = 2048;   // NUM_KERNELS

  float*          deg  = (float*)d_ws;
  float*          dvec = (float*)((char*)d_ws + 16384);
  unsigned short* Hbf  = (unsigned short*)((char*)d_ws + 32768);

  // normalization vector d = rsqrt(col-sums of ew)
  GNN_deg_init<<<n / 256, 256, 0, stream>>>(deg, n);
  GNN_deg_partial<<<dim3(n / 256, n / 128), 256, 0, stream>>>(ew, deg, n);
  GNN_deg_finalize<<<n / 256, 256, 0, stream>>>(deg, dvec, n);

  // h = x @ W (M=n, N=F, K=F), kept as bf16
  GNN_gemm_xw<<<dim3(F / 128, n / 128), 256, 0, stream>>>(x, W, Hbf, n, F, F);

  // out = diag(d) * ew^T*diag(d) * h + b (M=n, N=F, K=n)
  GNN_gemm_agg<<<dim3(F / 128, n / 128), 256, 0, stream>>>(ew, Hbf, dvec, b,
                                                           out, n, F);
}
